// MixerBlock_53102975647882
// MI455X (gfx1250) — matrix operand order repaired and checked by reading, not compiled, not run on hardware
//
#include <hip/hip_runtime.h>
#include <math.h>

// ---------------------------------------------------------------------------
// MI455X (gfx1250, wave32) MixerBlock.
// All GEMMs run through v_wmma_f32_16x16x32_bf16 (bf16 inputs, f32 accum).
// All B operands are pre-transposed to [N][K] so every global->LDS tile fill
// is a contiguous, 16B-aligned async b128 copy (GLOBAL_LOAD_ASYNC_TO_LDS_B128,
// tracked by ASYNCcnt and fenced with s_wait_asynccnt before the barrier).
// ---------------------------------------------------------------------------

typedef __bf16 bf16;
typedef __attribute__((ext_vector_type(16))) __bf16 v16bf;
typedef __attribute__((ext_vector_type(8)))  float  v8f;
typedef __attribute__((__vector_size__(16))) int    i32x4;

#define B_   64
#define P_   196
#define D_   512
#define HT_  7
#define HS_  16
#define HDT  35
#define HDS  40
#define INT_T 245
#define INT_S 640
#define HID_ 2048
#define PLD  200      // padded P leading dim (mult of 8 -> 16B aligned bf16 rows)
#define QKVT_LD 736   // padded 3*INT_T
#define QKVS_LD 1920
#define OCT_LD 248    // padded INT_T
#define HPAD 64       // padded head dim
#define BCH  8        // attention batch chunk
#define LDST 40       // LDS row stride (bf16 elems) -> 80B rows, 16B aligned

// --- CDNA5 async global->LDS copy (16 bytes), with guarded fallbacks -------
// The builtin takes typed int4 pointers: (global i32x4*, lds i32x4*, Ii, Ii).
// Global generic addresses are numerically addrspace(1); for LDS the generic
// address' low 32 bits are the LDS byte offset (ISA 10.2 aperture rules), so
// integer casts give valid addrspace pointers without addrspacecast.
#if defined(__AMDGCN__)
  #if __has_builtin(__builtin_amdgcn_global_load_async_to_lds_b128)
    #define ASYNC_CP16(gp, lp)                                                 \
      __builtin_amdgcn_global_load_async_to_lds_b128(                          \
          (__attribute__((address_space(1))) i32x4*)(unsigned long long)(gp),  \
          (__attribute__((address_space(3))) i32x4*)(unsigned)(unsigned long long)(lp), \
          0, 0)
  #else
    #define ASYNC_CP16(gp, lp)                                                 \
      asm volatile("global_load_async_to_lds_b128 %0, %1, off"                 \
                   :: "v"((unsigned)(unsigned long long)(lp)),                 \
                      "v"((unsigned long long)(gp)) : "memory")
  #endif
  #if __has_builtin(__builtin_amdgcn_s_wait_asynccnt)
    #define WAIT_ASYNC() __builtin_amdgcn_s_wait_asynccnt(0)
  #else
    #define WAIT_ASYNC() asm volatile("s_wait_asynccnt 0x0" ::: "memory")
  #endif
#else
  #define ASYNC_CP16(gp, lp) do { (void)(gp); (void)(lp); } while (0)
  #define WAIT_ASYNC()       do { } while (0)
#endif

// ---------------------------------------------------------------------------
// bf16 WMMA GEMM: C[MxN] = act(alpha * A[MxK] * B^T + bias + resid)
// A row-major [M][K] (lda), B row-major [N][K] (ldb) -- i.e. "transB" always.
// Block = 128 threads (4 waves), 64x64 tile, each wave a 32x32 sub-tile.
// ---------------------------------------------------------------------------
__global__ __launch_bounds__(128)
void gemm_bf16_kernel(const bf16* __restrict__ A, const bf16* __restrict__ Bm,
                      float* __restrict__ C, bf16* __restrict__ Cb,
                      const float* __restrict__ bias, const float* __restrict__ resid,
                      int M, int N, int K,
                      int lda, int ldb, int ldc, int ldcb,
                      long long strA, long long strB, long long strC, long long strCb,
                      float alpha, int act)
{
    __shared__ bf16 sA[64 * LDST];   // [m][k]
    __shared__ bf16 sB[64 * LDST];   // [n][k]

    const long long z = blockIdx.z;
    A += z * strA;
    Bm += z * strB;
    if (C)     C     += z * strC;
    if (Cb)    Cb    += z * strCb;
    if (resid) resid += z * strC;

    const int tid  = threadIdx.x;
    const int lane = tid & 31;
    const int wave = tid >> 5;
    const int wm = (wave >> 1) * 32;
    const int wn = (wave & 1) * 32;
    const int tM = blockIdx.y * 64;
    const int tN = blockIdx.x * 64;

    const int r  = tid >> 1;   // 0..63 (tile row)
    const int hf = tid & 1;    // 0/1  (16-elem K segment)
    const int frow = lane & 15;
    const int fhal = lane >> 4;

    v8f acc[2][2] = {};

    for (int k0 = 0; k0 < K; k0 += 32) {
        // ---- A tile (64 x 32): async b128 copies on the aligned fast path --
        {
            const int gr = tM + r;
            const int gk = k0 + hf * 16;
            bf16* dst = &sA[r * LDST + hf * 16];
            if (gr < M && gk + 16 <= K) {
                const bf16* src = A + (long long)gr * lda + gk;
                ASYNC_CP16(src, dst);
                ASYNC_CP16(src + 8, dst + 8);
                if (k0 + 32 < K)
                    __builtin_prefetch((const void*)(src + 32), 0, 0);
            } else {
                #pragma unroll
                for (int j = 0; j < 16; ++j) {
                    int kk = gk + j;
                    dst[j] = (gr < M && kk < K) ? A[(long long)gr * lda + kk] : (bf16)0.0f;
                }
            }
        }
        // ---- B tile (64 x 32), B stored [N][K] ----
        {
            const int gn = tN + r;
            const int gk = k0 + hf * 16;
            bf16* dst = &sB[r * LDST + hf * 16];
            if (gn < N && gk + 16 <= K) {
                const bf16* src = Bm + (long long)gn * ldb + gk;
                ASYNC_CP16(src, dst);
                ASYNC_CP16(src + 8, dst + 8);
                if (k0 + 32 < K)
                    __builtin_prefetch((const void*)(src + 32), 0, 0);
            } else {
                #pragma unroll
                for (int j = 0; j < 16; ++j) {
                    int kk = gk + j;
                    dst[j] = (gn < N && kk < K) ? Bm[(long long)gn * ldb + kk] : (bf16)0.0f;
                }
            }
        }
        WAIT_ASYNC();
        __syncthreads();

        // ---- fragments (layouts per CDNA5 ISA 7.12.2) ----
        v16bf af[2], bfv[2];
        #pragma unroll
        for (int i = 0; i < 2; ++i) {
            const bf16* base = &sA[(wm + i * 16 + frow) * LDST];
            union { uint4 u[2]; v16bf v; } t;
            t.u[0] = *(const uint4*)(base + fhal * 8);        // K = half*8 .. +7
            t.u[1] = *(const uint4*)(base + 16 + fhal * 8);   // K = 16+half*8 .. +7
            af[i] = t.v;
        }
        #pragma unroll
        for (int i = 0; i < 2; ++i) {
            const bf16* base = &sB[(wn + i * 16 + frow) * LDST];
            union { uint4 u[2]; v16bf v; } t;
            t.u[0] = *(const uint4*)(base + fhal * 8);        // K = half*8 .. +7
            t.u[1] = *(const uint4*)(base + 16 + fhal * 8);   // K = 16+half*8 .. +7
            bfv[i] = t.v;
        }
        #pragma unroll
        for (int i = 0; i < 2; ++i)
            #pragma unroll
            for (int j = 0; j < 2; ++j)
                acc[i][j] = __builtin_amdgcn_wmma_f32_16x16x32_bf16(
                    false, af[i], false, bfv[j], (short)0, acc[i][j], false, false);
        __syncthreads();
    }

    // ---- epilogue (C/D layout: N = lane&15, M = vgpr + (lane>>4)*8) ----
    #pragma unroll
    for (int i = 0; i < 2; ++i) {
        #pragma unroll
        for (int j = 0; j < 2; ++j) {
            const int gn = tN + wn + j * 16 + frow;
            if (gn >= N) continue;
            const float bv = bias ? bias[gn] : 0.0f;
            #pragma unroll
            for (int rr = 0; rr < 8; ++rr) {
                const int gm = tM + wm + i * 16 + fhal * 8 + rr;
                if (gm >= M) continue;
                float v = acc[i][j][rr] * alpha + bv;
                if (resid) v += resid[(long long)gm * ldc + gn];
                if (act == 1) v = 0.5f * v * (1.0f + erff(v * 0.70710678118f));
                if (C)  C [(long long)gm * ldc  + gn] = v;
                if (Cb) Cb[(long long)gm * ldcb + gn] = (bf16)v;
            }
        }
    }
}

// ---------------------------------------------------------------------------
// LayerNorm over D; optional transposed bf16 output (for token mixing: [B,D,P])
// ---------------------------------------------------------------------------
__global__ __launch_bounds__(256)
void layernorm_kernel(const float* __restrict__ x, const float* __restrict__ g,
                      const float* __restrict__ bta, bf16* __restrict__ out,
                      int D, int ldo, int transposed, int P)
{
    const int row = blockIdx.x;                 // b*P + p
    const float* xr = x + (long long)row * D;
    __shared__ float red[256];
    float s = 0.f;
    for (int d = threadIdx.x; d < D; d += 256) s += xr[d];
    red[threadIdx.x] = s; __syncthreads();
    for (int st = 128; st; st >>= 1) { if (threadIdx.x < st) red[threadIdx.x] += red[threadIdx.x + st]; __syncthreads(); }
    const float mean = red[0] / D;
    __syncthreads();
    float q = 0.f;
    for (int d = threadIdx.x; d < D; d += 256) { float t = xr[d] - mean; q += t * t; }
    red[threadIdx.x] = q; __syncthreads();
    for (int st = 128; st; st >>= 1) { if (threadIdx.x < st) red[threadIdx.x] += red[threadIdx.x + st]; __syncthreads(); }
    const float rstd = rsqrtf(red[0] / D + 1e-5f);
    for (int d = threadIdx.x; d < D; d += 256) {
        float v = (xr[d] - mean) * rstd * g[d] + bta[d];
        if (transposed) {
            int b = row / P, p = row % P;
            out[((long long)(b * D + d)) * ldo + p] = (bf16)v;
        } else {
            out[(long long)row * ldo + d] = (bf16)v;
        }
    }
}

// xc = xt - mean_over_P(xt), one block per (b,d) row
__global__ __launch_bounds__(128)
void center_rows_kernel(const bf16* __restrict__ xt, bf16* __restrict__ xc, int P, int ld)
{
    const long long row = blockIdx.x;
    const bf16* src = xt + row * ld;
    bf16* dst = xc + row * ld;
    __shared__ float red[128];
    float s = 0.f;
    for (int p = threadIdx.x; p < P; p += 128) s += (float)src[p];
    red[threadIdx.x] = s; __syncthreads();
    for (int st = 64; st; st >>= 1) { if (threadIdx.x < st) red[threadIdx.x] += red[threadIdx.x + st]; __syncthreads(); }
    const float mean = red[0] / P;
    for (int p = threadIdx.x; p < P; p += 128) dst[p] = (bf16)((float)src[p] - mean);
}

// token softmax: bias = cov_scale[h] * cov[b] ; N = 512
__global__ __launch_bounds__(256)
void softmax_tok_kernel(const float* __restrict__ S, bf16* __restrict__ Pr,
                        const float* __restrict__ cov, const float* __restrict__ cov_scale,
                        int b0)
{
    const int N = D_;
    const int idx = blockIdx.x;
    const int i  = idx % N;
    const int h  = (idx / N) % HT_;
    const int bc = idx / (N * HT_);
    const float* srow = S   + (((long long)(bc * HT_ + h)) * N + i) * N;
    const float* crow = cov + (((long long)(b0 + bc)) * N + i) * N;
    bf16* orow = Pr + (((long long)(bc * HT_ + h)) * N + i) * N;
    const float scl = cov_scale[h];
    __shared__ float red[256];
    float m = -1e30f;
    for (int j = threadIdx.x; j < N; j += 256) m = fmaxf(m, srow[j] + scl * crow[j]);
    red[threadIdx.x] = m; __syncthreads();
    for (int st = 128; st; st >>= 1) { if (threadIdx.x < st) red[threadIdx.x] = fmaxf(red[threadIdx.x], red[threadIdx.x + st]); __syncthreads(); }
    m = red[0]; __syncthreads();
    float s = 0.f;
    for (int j = threadIdx.x; j < N; j += 256) s += __expf(srow[j] + scl * crow[j] - m);
    red[threadIdx.x] = s; __syncthreads();
    for (int st = 128; st; st >>= 1) { if (threadIdx.x < st) red[threadIdx.x] += red[threadIdx.x + st]; __syncthreads(); }
    const float inv = 1.0f / red[0];
    for (int j = threadIdx.x; j < N; j += 256)
        orow[j] = (bf16)(__expf(srow[j] + scl * crow[j] - m) * inv);
}

// seq softmax: bias = relpos bias [h][i][j]; rows padded to PLD
__global__ __launch_bounds__(256)
void softmax_seq_kernel(const float* __restrict__ S, bf16* __restrict__ Pr,
                        const float* __restrict__ sbias)
{
    const int N = P_;
    const int idx = blockIdx.x;
    const int i  = idx % N;
    const int h  = (idx / N) % HS_;
    const int bc = idx / (N * HS_);
    const float* srow = S + (((long long)(bc * HS_ + h)) * N + i) * PLD;
    const float* brow = sbias + ((long long)h * N + i) * N;
    bf16* orow = Pr + (((long long)(bc * HS_ + h)) * N + i) * PLD;
    __shared__ float red[256];
    float m = -1e30f;
    for (int j = threadIdx.x; j < N; j += 256) m = fmaxf(m, srow[j] + brow[j]);
    red[threadIdx.x] = m; __syncthreads();
    for (int st = 128; st; st >>= 1) { if (threadIdx.x < st) red[threadIdx.x] = fmaxf(red[threadIdx.x], red[threadIdx.x + st]); __syncthreads(); }
    m = red[0]; __syncthreads();
    float s = 0.f;
    for (int j = threadIdx.x; j < N; j += 256) s += __expf(srow[j] + brow[j] - m);
    red[threadIdx.x] = s; __syncthreads();
    for (int st = 128; st; st >>= 1) { if (threadIdx.x < st) red[threadIdx.x] += red[threadIdx.x + st]; __syncthreads(); }
    const float inv = 1.0f / red[0];
    for (int j = threadIdx.x; j < N; j += 256)
        orow[j] = (bf16)(__expf(srow[j] + brow[j] - m) * inv);
}

// split heads + pad to hd=64, f32 -> bf16.
// Q,K: [b][h][n][64] (A and B^T operands of the score GEMM).
// V:   [b][h][e(64)][n] transposed with leading dim vld (B^T operand of AV).
__global__ __launch_bounds__(256)
void pack_qkv_kernel(const float* __restrict__ qkv, int ldq, int inner, int H, int hd, int N,
                     int vld,
                     bf16* __restrict__ Q, bf16* __restrict__ K, bf16* __restrict__ V)
{
    long long t = (long long)blockIdx.x * 256 + threadIdx.x;
    const long long total = (long long)B_ * H * N * HPAD;
    if (t >= total) return;
    const int e = (int)(t & (HPAD - 1));
    long long r = t >> 6;
    const int n = (int)(r % N); r /= N;
    const int h = (int)(r % H);
    const int b = (int)(r / H);
    float q = 0.f, k = 0.f, v = 0.f;
    if (e < hd) {
        const float* base = qkv + ((long long)(b * N + n)) * ldq + h * hd + e;
        q = base[0]; k = base[inner]; v = base[2 * inner];
    }
    Q[t] = (bf16)q; K[t] = (bf16)k;
    V[(((long long)(b * H + h)) * HPAD + e) * vld + n] = (bf16)v;
}

// merge heads: O[b][h][n][64] f32 -> Oc[b][n][h*hd+e] bf16
__global__ __launch_bounds__(256)
void merge_heads_kernel(const float* __restrict__ O, bf16* __restrict__ Oc,
                        int H, int hd, int N, int inner, int ldo)
{
    long long t = (long long)blockIdx.x * 256 + threadIdx.x;
    const long long total = (long long)B_ * N * inner;
    if (t >= total) return;
    const int c = (int)(t % inner);
    long long r = t / inner;
    const int n = (int)(r % N);
    const int b = (int)(r / N);
    const int h = c / hd, e = c % hd;
    Oc[((long long)(b * N + n)) * ldo + c] =
        (bf16)O[(((long long)(b * H + h)) * N + n) * HPAD + e];
}

// x1[b][p][d] = x[b][p][d] + y[b][d][p]   (y has leading dim PLD)
__global__ __launch_bounds__(256)
void add_transpose_kernel(const float* __restrict__ x, const float* __restrict__ y,
                          float* __restrict__ x1)
{
    long long t = (long long)blockIdx.x * 256 + threadIdx.x;
    const long long total = (long long)B_ * P_ * D_;
    if (t >= total) return;
    const int d = (int)(t % D_);
    long long r = t / D_;
    const int p = (int)(r % P_);
    const int b = (int)(r / P_);
    x1[t] = x[t] + y[((long long)(b * D_ + d)) * PLD + p];
}

// sbias[h][i][j] = relpos_table[rel_idx[i][j]][h]
__global__ __launch_bounds__(256)
void relpos_bias_kernel(const float* __restrict__ table, const int* __restrict__ idx,
                        float* __restrict__ out)
{
    long long t = (long long)blockIdx.x * 256 + threadIdx.x;
    const long long total = (long long)HS_ * P_ * P_;
    if (t >= total) return;
    const int j = (int)(t % P_);
    long long r = t / P_;
    const int i = (int)(r % P_);
    const int h = (int)(r / P_);
    out[t] = table[(long long)idx[i * P_ + j] * HS_ + h];
}

// Weight convert f32 [rows][cols] -> bf16 TRANSPOSED [cols][ldo] (ldo >= rows).
__global__ __launch_bounds__(256)
void convert_w_kernel(const float* __restrict__ w, bf16* __restrict__ o,
                      int rows, int cols, int ldo)
{
    long long t = (long long)blockIdx.x * 256 + threadIdx.x;
    if (t >= (long long)rows * cols) return;
    const int rr = (int)(t / cols), cc = (int)(t % cols);
    o[(long long)cc * ldo + rr] = (bf16)w[t];
}

// ---------------------------------------------------------------------------
static inline int ceildiv(int a, int b) { return (a + b - 1) / b; }

static void gemm(hipStream_t st, const bf16* A, const bf16* Bm, float* C, bf16* Cb,
                 const float* bias, const float* resid, int M, int N, int K,
                 int lda, int ldb, int ldc, int ldcb,
                 long long sA, long long sB, long long sC, long long sCb,
                 int batch, float alpha, int act)
{
    dim3 g(ceildiv(N, 64), ceildiv(M, 64), batch), b(128, 1, 1);
    gemm_bf16_kernel<<<g, b, 0, st>>>(A, Bm, C, Cb, bias, resid, M, N, K,
                                      lda, ldb, ldc, ldcb, sA, sB, sC, sCb,
                                      alpha, act);
}

extern "C" void kernel_launch(void* const* d_in, const int* in_sizes, int n_in,
                              void* d_out, int out_size, void* d_ws, size_t ws_size,
                              hipStream_t stream)
{
    const float* x         = (const float*)d_in[0];
    const float* tok_g     = (const float*)d_in[1];
    const float* tok_bb    = (const float*)d_in[2];
    const float* tok_qkv_w = (const float*)d_in[3];
    const float* tok_qkv_b = (const float*)d_in[4];
    const float* tok_out_w = (const float*)d_in[5];
    const float* tok_out_b = (const float*)d_in[6];
    const float* cov_scale = (const float*)d_in[7];
    const float* seq_g     = (const float*)d_in[8];
    const float* seq_bb    = (const float*)d_in[9];
    const float* seq_qkv_w = (const float*)d_in[10];
    const float* seq_qkv_b = (const float*)d_in[11];
    const float* seq_out_w = (const float*)d_in[12];
    const float* seq_out_b = (const float*)d_in[13];
    const float* relpos    = (const float*)d_in[14];
    const float* mlp_g     = (const float*)d_in[15];
    const float* mlp_bbias = (const float*)d_in[16];
    const float* mlp_w1    = (const float*)d_in[17];
    const float* mlp_b1    = (const float*)d_in[18];
    const float* mlp_w2    = (const float*)d_in[19];
    const float* mlp_b2    = (const float*)d_in[20];
    const int*   rel_idx   = (const int*)d_in[21];
    float* out = (float*)d_out;
    (void)in_sizes; (void)n_in; (void)out_size; (void)ws_size;

    char* ws = (char*)d_ws;
    size_t off = 0;
    auto alloc = [&](size_t bytes) -> char* {
        off = (off + 255) & ~(size_t)255;
        char* p = ws + off; off += bytes; return p;
    };

    // Weights stored transposed [N][K] with padded leading dims.
    bf16*  wt_tq = (bf16*)alloc((size_t)QKVT_LD * PLD * 2);     // [735][200]
    bf16*  wt_to = (bf16*)alloc((size_t)P_ * OCT_LD * 2);       // [196][248]
    bf16*  wt_sq = (bf16*)alloc((size_t)QKVS_LD * D_ * 2);      // [1920][512]
    bf16*  wt_so = (bf16*)alloc((size_t)D_ * INT_S * 2);        // [512][640]
    bf16*  wt_m1 = (bf16*)alloc((size_t)HID_ * D_ * 2);         // [2048][512]
    bf16*  wt_m2 = (bf16*)alloc((size_t)D_ * HID_ * 2);         // [512][2048]
    bf16*  xt    = (bf16*)alloc((size_t)B_ * D_ * PLD * 2);
    bf16*  xc    = (bf16*)alloc((size_t)B_ * D_ * PLD * 2);
    bf16*  lnb   = (bf16*)alloc((size_t)B_ * P_ * D_ * 2);
    float* cov   = (float*)alloc((size_t)B_ * D_ * D_ * 4);
    bf16*  hbuf  = (bf16*)cov;   // overlay: MLP hidden lives after cov is dead
    float* qkvf  = (float*)alloc((size_t)B_ * D_ * QKVT_LD * 4);  // >= B*P*1920
    bf16*  Qp    = (bf16*)alloc((size_t)B_ * HT_ * D_ * HPAD * 2);
    bf16*  Kp    = (bf16*)alloc((size_t)B_ * HT_ * D_ * HPAD * 2);
    bf16*  Vt    = (bf16*)alloc((size_t)B_ * HT_ * HPAD * D_ * 2); // transposed V
    float* sc    = (float*)alloc((size_t)BCH * HT_ * D_ * D_ * 4);
    bf16*  pr    = (bf16*)alloc((size_t)BCH * HT_ * D_ * D_ * 2);
    float* Obuf  = (float*)alloc((size_t)B_ * HT_ * D_ * HPAD * 4);
    bf16*  Oc    = (bf16*)alloc((size_t)B_ * D_ * OCT_LD * 2);    // >= B*196*640
    float* ybuf  = (float*)alloc((size_t)B_ * D_ * PLD * 4);
    float* x1    = (float*)alloc((size_t)B_ * P_ * D_ * 4);
    float* x2    = (float*)alloc((size_t)B_ * P_ * D_ * 4);
    float* sbias = (float*)alloc((size_t)HS_ * P_ * P_ * 4);

    dim3 t256(256);
    auto grid1 = [](long long n) { return dim3((unsigned)((n + 255) / 256)); };

    // weight conversions (transposed) + relpos bias
    convert_w_kernel<<<grid1((long long)P_ * 3 * INT_T), t256, 0, stream>>>(tok_qkv_w, wt_tq, P_, 3 * INT_T, PLD);
    convert_w_kernel<<<grid1((long long)INT_T * P_), t256, 0, stream>>>(tok_out_w, wt_to, INT_T, P_, OCT_LD);
    convert_w_kernel<<<grid1((long long)D_ * QKVS_LD), t256, 0, stream>>>(seq_qkv_w, wt_sq, D_, 3 * INT_S, D_);
    convert_w_kernel<<<grid1((long long)INT_S * D_), t256, 0, stream>>>(seq_out_w, wt_so, INT_S, D_, INT_S);
    convert_w_kernel<<<grid1((long long)D_ * HID_), t256, 0, stream>>>(mlp_w1, wt_m1, D_, HID_, D_);
    convert_w_kernel<<<grid1((long long)HID_ * D_), t256, 0, stream>>>(mlp_w2, wt_m2, HID_, D_, HID_);
    relpos_bias_kernel<<<grid1((long long)HS_ * P_ * P_), t256, 0, stream>>>(relpos, rel_idx, sbias);

    // ---------------- token mixing ----------------
    layernorm_kernel<<<B_ * P_, 256, 0, stream>>>(x, tok_g, tok_bb, xt, D_, PLD, 1, P_);
    center_rows_kernel<<<B_ * D_, 128, 0, stream>>>(xt, xc, P_, PLD);
    // cov[b] = xc xc^T / P   (batched; B operand = xc itself, already [N][K])
    gemm(stream, xc, xc, cov, nullptr, nullptr, nullptr, D_, D_, P_,
         PLD, PLD, D_, 0, (long long)D_ * PLD, (long long)D_ * PLD, (long long)D_ * D_, 0,
         B_, 1.0f / (float)P_, 0);
    // token qkv (flat M = B*D)
    gemm(stream, xt, wt_tq, qkvf, nullptr, tok_qkv_b, nullptr, B_ * D_, 3 * INT_T, P_,
         PLD, PLD, QKVT_LD, 0, 0, 0, 0, 0, 1, 1.0f, 0);
    pack_qkv_kernel<<<grid1((long long)B_ * HT_ * D_ * HPAD), t256, 0, stream>>>(
        qkvf, QKVT_LD, INT_T, HT_, HDT, D_, D_, Qp, Kp, Vt);
    const float at = 1.0f / sqrtf((float)HDT);
    for (int b0 = 0; b0 < B_; b0 += BCH) {
        const long long hb  = (long long)b0 * HT_ * D_ * HPAD;
        gemm(stream, Qp + hb, Kp + hb, sc, nullptr, nullptr, nullptr, D_, D_, HPAD,
             HPAD, HPAD, D_, 0, (long long)D_ * HPAD, (long long)D_ * HPAD, (long long)D_ * D_, 0,
             BCH * HT_, at, 0);
        softmax_tok_kernel<<<BCH * HT_ * D_, 256, 0, stream>>>(sc, pr, cov, cov_scale, b0);
        gemm(stream, pr, Vt + hb, Obuf + hb, nullptr, nullptr, nullptr, D_, HPAD, D_,
             D_, D_, HPAD, 0, (long long)D_ * D_, (long long)HPAD * D_, (long long)D_ * HPAD, 0,
             BCH * HT_, 1.0f, 0);
    }
    merge_heads_kernel<<<grid1((long long)B_ * D_ * INT_T), t256, 0, stream>>>(
        Obuf, Oc, HT_, HDT, D_, INT_T, OCT_LD);
    gemm(stream, Oc, wt_to, ybuf, nullptr, tok_out_b, nullptr, B_ * D_, P_, INT_T,
         OCT_LD, OCT_LD, PLD, 0, 0, 0, 0, 0, 1, 1.0f, 0);
    add_transpose_kernel<<<grid1((long long)B_ * P_ * D_), t256, 0, stream>>>(x, ybuf, x1);

    // ---------------- seq mixing ----------------
    layernorm_kernel<<<B_ * P_, 256, 0, stream>>>(x1, seq_g, seq_bb, lnb, D_, D_, 0, P_);
    gemm(stream, lnb, wt_sq, qkvf, nullptr, seq_qkv_b, nullptr, B_ * P_, 3 * INT_S, D_,
         D_, D_, QKVS_LD, 0, 0, 0, 0, 0, 1, 1.0f, 0);
    pack_qkv_kernel<<<grid1((long long)B_ * HS_ * P_ * HPAD), t256, 0, stream>>>(
        qkvf, QKVS_LD, INT_S, HS_, HDS, P_, PLD, Qp, Kp, Vt);
    const float as = 1.0f / sqrtf((float)HDS);
    for (int b0 = 0; b0 < B_; b0 += BCH) {
        const long long hb  = (long long)b0 * HS_ * P_ * HPAD;
        const long long hbv = (long long)b0 * HS_ * HPAD * PLD;
        gemm(stream, Qp + hb, Kp + hb, sc, nullptr, nullptr, nullptr, P_, P_, HPAD,
             HPAD, HPAD, PLD, 0, (long long)P_ * HPAD, (long long)P_ * HPAD, (long long)P_ * PLD, 0,
             BCH * HS_, as, 0);
        softmax_seq_kernel<<<BCH * HS_ * P_, 256, 0, stream>>>(sc, pr, sbias);
        gemm(stream, pr, Vt + hbv, Obuf + hb, nullptr, nullptr, nullptr, P_, HPAD, P_,
             PLD, PLD, HPAD, 0, (long long)P_ * PLD, (long long)HPAD * PLD, (long long)P_ * HPAD, 0,
             BCH * HS_, 1.0f, 0);
    }
    merge_heads_kernel<<<grid1((long long)B_ * P_ * INT_S), t256, 0, stream>>>(
        Obuf, Oc, HS_, HDS, P_, INT_S, INT_S);
    gemm(stream, Oc, wt_so, x2, nullptr, seq_out_b, x1, B_ * P_, D_, INT_S,
         INT_S, INT_S, D_, 0, 0, 0, 0, 0, 1, 1.0f, 0);

    // ---------------- MLP ----------------
    layernorm_kernel<<<B_ * P_, 256, 0, stream>>>(x2, mlp_g, mlp_bbias, lnb, D_, D_, 0, P_);
    gemm(stream, lnb, wt_m1, nullptr, hbuf, mlp_b1, nullptr, B_ * P_, HID_, D_,
         D_, D_, 0, HID_, 0, 0, 0, 0, 1, 1.0f, 1);              // GELU, bf16 out
    gemm(stream, hbuf, wt_m2, out, nullptr, mlp_b2, x2, B_ * P_, D_, HID_,
         HID_, HID_, D_, 0, 0, 0, 0, 0, 1, 1.0f, 0);            // + residual -> d_out
}
